// NeuralRDE_10015863734766
// MI455X (gfx1250) — matrix operand order repaired and checked by reading, not compiled, not run on hardware
//
#include <hip/hip_runtime.h>

typedef unsigned short u16;
typedef unsigned int   u32;
typedef __bf16 v16bf __attribute__((ext_vector_type(16)));
typedef float  v8f   __attribute__((ext_vector_type(8)));
typedef u32    u32x4 __attribute__((ext_vector_type(4)));
typedef int    v4i   __attribute__((ext_vector_type(4)));

union ABu { v16bf v; u32x4 q[2]; };

#define T_LEN 4096
#define D_IN  8
#define HID   512
#define LSIG  36
#define HL    (HID * LSIG)               // 18432
#define NWG   16                         // one cluster: 16 workgroups, one per WGP
#define NSTEP (T_LEN - 1)                // 4095 scan steps
#define ROWS_PER_WG  (HL / NWG)          // 1152 Wf3 output rows per WGP
#define TILES_PER_WG (ROWS_PER_WG / 16)  // 72 WMMA row-tiles per WGP
#define PAIRS_PER_WG (TILES_PER_WG / 2)  // 36 tile-pairs per WGP

#if defined(__has_builtin)
#if __has_builtin(__builtin_amdgcn_global_load_async_to_lds_b128) && \
    __has_builtin(__builtin_amdgcn_s_wait_asynccnt)
#define HAVE_ASYNC_LDS 1
#endif
#endif

// ---------- helpers ----------
__device__ __forceinline__ u16 f2bf(float f) {        // fp32 -> bf16 RNE
  u32 u = __float_as_uint(f);
  u += 0x7FFFu + ((u >> 16) & 1u);
  return (u16)(u >> 16);
}
__device__ __forceinline__ float softplus_fast(float x) {
  return (x > 20.0f) ? x : __logf(1.0f + __expf(x));
}
__device__ __forceinline__ float tanh_fast(float x) {
  float e = __expf(2.0f * x);             // +inf / 0 at extremes -> +/-1, correct
  return 1.0f - 2.0f / (e + 1.0f);
}

// Cluster-wide sync: fast path = CDNA5 cluster barrier (s_barrier_signal/wait -3),
// functional path = monotonic atomic counter (NOP-safe if not dispatched as cluster).
__device__ __forceinline__ void cluster_sync(unsigned* cnt, unsigned& target) {
  __threadfence();
  __syncthreads();
  target += gridDim.x;
  if (threadIdx.x == 0) {
    atomicAdd(cnt, 1u);
    while (__hip_atomic_load(cnt, __ATOMIC_ACQUIRE, __HIP_MEMORY_SCOPE_AGENT) < target)
      __builtin_amdgcn_s_sleep(1);
  }
  __builtin_amdgcn_s_cluster_barrier();   // s_barrier_signal -3 ; s_barrier_wait -3
  __syncthreads();
  __threadfence();
}

// Dual 16-row x K=512 GEMV tiles on the matrix pipe. Two independent A streams
// share one broadcast B fragment: 2 WMMAs per load-wait, 2x loads in flight,
// half the LDS traffic. B = vector broadcast into all 16 columns.
__device__ __forceinline__ void wmma_gemv16x2(const u16* __restrict__ wrow0,
                                              const u16* __restrict__ wrow1,
                                              const u16* __restrict__ xbf, int lane,
                                              v8f& acc0, v8f& acc1) {
  const int s0 = (lane < 16) ? 0 : 16;    // A-frag K chunks {0,8} / {16,24}
  const int bo = (lane & 16) ? 16 : 0;    // B-frag K half
  #pragma unroll 4
  for (int k0 = 0; k0 < HID; k0 += 32) {
    ABu a0, a1, b;
    b.v     = *(const v16bf*)(xbf + k0 + bo);
    a0.q[0] = *(const u32x4*)(wrow0 + k0 + s0);
    a0.q[1] = *(const u32x4*)(wrow0 + k0 + s0 + 8);
    a1.q[0] = *(const u32x4*)(wrow1 + k0 + s0);
    a1.q[1] = *(const u32x4*)(wrow1 + k0 + s0 + 8);
    __builtin_prefetch((const void*)(wrow0 + k0 + 256), 0, 3);  // unconditional,
    __builtin_prefetch((const void*)(wrow1 + k0 + 256), 0, 3);  // speculative
    acc0 = __builtin_amdgcn_wmma_f32_16x16x32_bf16(false, a0.v, false, b.v,
                                                   (short)0, acc0, false, false);
    acc1 = __builtin_amdgcn_wmma_f32_16x16x32_bf16(false, a1.v, false, b.v,
                                                   (short)0, acc1, false, false);
  }
}

// ---------- kernels ----------
__global__ void cvt_bf16(const float* __restrict__ src, u16* __restrict__ dst, int n) {
  int i = blockIdx.x * blockDim.x + threadIdx.x;
  if (i < n) dst[i] = f2bf(src[i]);
}

__global__ void zero_u32(unsigned* p) { *p = 0u; }

// log-signature features, one thread per window t (parallel over 4095 windows)
__global__ void logsig_kernel(const float* __restrict__ values, float* __restrict__ out) {
  int t = blockIdx.x * blockDim.x + threadIdx.x;
  if (t >= NSTEP) return;
  float P[D_IN];  float l2[28];
  #pragma unroll
  for (int d = 0; d < D_IN; ++d) P[d] = 0.f;
  #pragma unroll
  for (int p = 0; p < 28; ++p) l2[p] = 0.f;
  for (int w = 0; w < 8; ++w) {
    int i0 = t + w;     if (i0 > T_LEN - 1) i0 = T_LEN - 1;
    int i1 = t + w + 1; if (i1 > T_LEN - 1) i1 = T_LEN - 1;
    float dw[D_IN];
    #pragma unroll
    for (int d = 0; d < D_IN; ++d)
      dw[d] = values[(size_t)i1 * D_IN + d] - values[(size_t)i0 * D_IN + d];
    int p = 0;
    #pragma unroll
    for (int i = 0; i < D_IN; ++i)
      #pragma unroll
      for (int j = i + 1; j < D_IN; ++j) { l2[p] += P[i] * dw[j] - P[j] * dw[i]; ++p; }
    #pragma unroll
    for (int d = 0; d < D_IN; ++d) P[d] += dw[d];
  }
  float* o = out + (size_t)t * LSIG;
  #pragma unroll
  for (int d = 0; d < D_IN; ++d) o[d] = P[d];               // lvl1
  #pragma unroll
  for (int p = 0; p < 28; ++p) o[D_IN + p] = 0.5f * l2[p];  // lvl2 (triu order)
}

// init MLP -> h0 (tiny, one workgroup)
__global__ void __launch_bounds__(HID)
init_h0(const float* __restrict__ values,
        const float* __restrict__ Wi1, const float* __restrict__ bi1,
        const float* __restrict__ Wi2, const float* __restrict__ bi2,
        const float* __restrict__ Wi3, const float* __restrict__ bi3,
        float* __restrict__ hiddenF, u16* __restrict__ hiddenBF) {
  __shared__ float s1[HID];
  __shared__ float s2[HID];
  const int r = threadIdx.x;
  float a = bi1[r];
  #pragma unroll
  for (int d = 0; d < D_IN; ++d) a += Wi1[r * D_IN + d] * values[d];
  s1[r] = softplus_fast(a);
  __syncthreads();
  float b = bi2[r];
  for (int k = 0; k < HID; ++k) b += Wi2[(size_t)r * HID + k] * s1[k];
  s2[r] = softplus_fast(b);
  __syncthreads();
  float c = bi3[r];
  for (int k = 0; k < HID; ++k) c += Wi3[(size_t)r * HID + k] * s2[k];
  hiddenF[r]  = c;
  hiddenBF[r] = f2bf(c);
}

// persistent sequential scan: 16 WGs (one per WGP, one cluster) x 512 threads.
// GEMV1/GEMV2 computed redundantly per WGP (activations in LDS) so only ONE
// cluster-wide sync sits on the serial critical path per step.
__global__ void __launch_bounds__(512, 1)
scan_kernel(const u16* __restrict__ Wf1b, const u16* __restrict__ Wf2b,
            const u16* __restrict__ Wf3b,
            const float* __restrict__ bf1, const float* __restrict__ bf2,
            const float* __restrict__ bf3,
            const float* __restrict__ logsigs,
            float* __restrict__ hiddenF, u16* __restrict__ hiddenBF,
            unsigned* __restrict__ sync_cnt) {
  const int tid  = threadIdx.x;
  const int lane = tid & 31;
  const int wave = tid >> 5;
  const int wg   = blockIdx.x;

  __shared__ __align__(64) u16 h_bf[HID];   // current hidden state (bf16)
  __shared__ __align__(64) u16 a_bf[HID];   // softplus(Wf1 h + b)
  __shared__ __align__(64) u16 z_bf[HID];   // softplus(Wf2 a + b)
  __shared__ float smat[ROWS_PER_WG];       // this WGP's slice of tanh(Wf3 z + b)
  __shared__ float sls[LSIG];

  unsigned target = 0;
  const int lrow = lane & 15;               // A-frag row within tile
  const int mo   = (lane & 16) ? 8 : 0;     // C-frag row offset for this half-wave

  for (int t = 0; t < NSTEP; ++t) {
    // ---- broadcast h(t) into LDS (async global->LDS engine)
    const u16* hrow = hiddenBF + (size_t)t * HID;
#ifdef HAVE_ASYNC_LDS
    if (tid < 64) {   // 64 lanes x 16B = 1KB row
      __builtin_amdgcn_global_load_async_to_lds_b128(
          (__attribute__((address_space(1))) v4i*)(hrow + tid * 8),
          (__attribute__((address_space(3))) v4i*)(&h_bf[tid * 8]), 0, 0);
      __builtin_amdgcn_s_wait_asynccnt(0);
    }
#else
    if (tid < 256) ((u32*)h_bf)[tid] = ((const u32*)hrow)[tid];
#endif
    if (tid < LSIG) sls[tid] = logsigs[(size_t)t * LSIG + tid];
    __syncthreads();

    // ---- phase A: a = softplus(Wf1 @ h + bf1); tiles (wave, wave+16) per wave
    {
      const int rb0 = wave * 16, rb1 = (wave + 16) * 16;
      v8f acc0 = {}, acc1 = {};
      wmma_gemv16x2(Wf1b + (size_t)(rb0 + lrow) * HID,
                    Wf1b + (size_t)(rb1 + lrow) * HID, h_bf, lane, acc0, acc1);
      if ((lane & 15) == 0) {
        #pragma unroll
        for (int rr = 0; rr < 8; ++rr) {
          a_bf[rb0 + mo + rr] = f2bf(softplus_fast(acc0[rr] + bf1[rb0 + mo + rr]));
          a_bf[rb1 + mo + rr] = f2bf(softplus_fast(acc1[rr] + bf1[rb1 + mo + rr]));
        }
      }
    }
    __syncthreads();

    // ---- phase B: z = softplus(Wf2 @ a + bf2)
    {
      const int rb0 = wave * 16, rb1 = (wave + 16) * 16;
      v8f acc0 = {}, acc1 = {};
      wmma_gemv16x2(Wf2b + (size_t)(rb0 + lrow) * HID,
                    Wf2b + (size_t)(rb1 + lrow) * HID, a_bf, lane, acc0, acc1);
      if ((lane & 15) == 0) {
        #pragma unroll
        for (int rr = 0; rr < 8; ++rr) {
          z_bf[rb0 + mo + rr] = f2bf(softplus_fast(acc0[rr] + bf2[rb0 + mo + rr]));
          z_bf[rb1 + mo + rr] = f2bf(softplus_fast(acc1[rr] + bf2[rb1 + mo + rr]));
        }
      }
    }
    __syncthreads();

    // ---- phase C: this WGP's 1152 rows of tanh(Wf3 @ z + bf3), 36 tile-pairs
    for (int pp = wave; pp < PAIRS_PER_WG; pp += 16) {
      const int rbl0 = (2 * pp) * 16;                          // local row bases
      const int rbl1 = rbl0 + 16;
      const size_t rbg0 = (size_t)wg * ROWS_PER_WG + rbl0;     // global row bases
      const size_t rbg1 = rbg0 + 16;
      v8f acc0 = {}, acc1 = {};
      wmma_gemv16x2(Wf3b + (rbg0 + lrow) * HID,
                    Wf3b + (rbg1 + lrow) * HID, z_bf, lane, acc0, acc1);
      if ((lane & 15) == 0) {
        #pragma unroll
        for (int rr = 0; rr < 8; ++rr) {
          smat[rbl0 + mo + rr] = tanh_fast(acc0[rr] + bf3[rbg0 + mo + rr]);
          smat[rbl1 + mo + rr] = tanh_fast(acc1[rr] + bf3[rbg1 + mo + rr]);
        }
      }
    }
    __syncthreads();

    // ---- h(t+1) = h(t) + mat @ ls for this WGP's 32 hidden rows
    if (tid < 32) {
      const int gr = wg * 32 + tid;
      float s = hiddenF[(size_t)t * HID + gr];
      const float* mrow = &smat[tid * LSIG];
      #pragma unroll
      for (int j = 0; j < LSIG; ++j) s += mrow[j] * sls[j];
      hiddenF[(size_t)(t + 1) * HID + gr]  = s;
      hiddenBF[(size_t)(t + 1) * HID + gr] = f2bf(s);
    }

    // ---- single cluster-wide sync per step
    cluster_sync(sync_cnt, target);
  }
}

// epilogue: out = hidden(4096x512) @ Wr^T(512x10) + br, pure WMMA, tile per wave
__global__ void __launch_bounds__(256)
out_gemm(const u16* __restrict__ hiddenBF, const u16* __restrict__ WrBF,
         const float* __restrict__ br, float* __restrict__ out) {
  const int lane = threadIdx.x & 31;
  const int wave = threadIdx.x >> 5;
  const int tile = blockIdx.x * 8 + wave;          // 32 blocks x 8 waves = 256 tiles
  const int rowbase = tile * 16;
  const u16* arow = hiddenBF + (size_t)(rowbase + (lane & 15)) * HID;
  const u16* brow = WrBF + (size_t)(lane & 15) * HID;  // B col = lane&15 (cols>=10 unused)
  const int s0 = (lane < 16) ? 0 : 16;
  const int bo = (lane & 16) ? 16 : 0;
  v8f acc = {};
  #pragma unroll 4
  for (int k0 = 0; k0 < HID; k0 += 32) {
    ABu a, b;
    a.q[0] = *(const u32x4*)(arow + k0 + s0);
    a.q[1] = *(const u32x4*)(arow + k0 + s0 + 8);
    b.v    = *(const v16bf*)(brow + k0 + bo);
    acc = __builtin_amdgcn_wmma_f32_16x16x32_bf16(false, a.v, false, b.v,
                                                  (short)0, acc, false, false);
  }
  const int n = lane & 15;
  if (n < 10) {
    const int mb = rowbase + ((lane & 16) ? 8 : 0);
    const float bb = br[n];
    #pragma unroll
    for (int rr = 0; rr < 8; ++rr)
      out[(size_t)(mb + rr) * 10 + n] = acc[rr] + bb;
  }
}

// ---------- launch ----------
extern "C" void kernel_launch(void* const* d_in, const int* in_sizes, int n_in,
                              void* d_out, int out_size, void* d_ws, size_t ws_size,
                              hipStream_t stream) {
  (void)in_sizes; (void)n_in; (void)out_size; (void)ws_size;
  const float* values = (const float*)d_in[1];
  const float* Wi1 = (const float*)d_in[2];
  const float* bi1 = (const float*)d_in[3];
  const float* Wi2 = (const float*)d_in[4];
  const float* bi2 = (const float*)d_in[5];
  const float* Wi3 = (const float*)d_in[6];
  const float* bi3 = (const float*)d_in[7];
  const float* Wf1 = (const float*)d_in[8];
  const float* bf1 = (const float*)d_in[9];
  const float* Wf2 = (const float*)d_in[10];
  const float* bf2 = (const float*)d_in[11];
  const float* Wf3 = (const float*)d_in[12];
  const float* bf3 = (const float*)d_in[13];
  const float* Wr  = (const float*)d_in[14];
  const float* br  = (const float*)d_in[15];
  float* out = (float*)d_out;

  // carve workspace (256B aligned slots)
  char* p = (char*)d_ws;
  auto carve = [&](size_t bytes) -> char* {
    char* r = p; p += (bytes + 255) & ~(size_t)255; return r;
  };
  u16*   Wf1b     = (u16*)carve((size_t)HID * HID * 2);
  u16*   Wf2b     = (u16*)carve((size_t)HID * HID * 2);
  u16*   Wf3b     = (u16*)carve((size_t)HL * HID * 2);
  u16*   WrBF     = (u16*)carve((size_t)16 * HID * 2);   // only 10 rows used
  float* hiddenF  = (float*)carve((size_t)T_LEN * HID * 4);
  u16*   hiddenBF = (u16*)carve((size_t)T_LEN * HID * 2);
  float* logsigs  = (float*)carve((size_t)NSTEP * LSIG * 4);
  unsigned* cnt   = (unsigned*)carve(256);

  // 1) weight conversion to bf16 (WMMA operand precision; halves L2 traffic)
  cvt_bf16<<<(HID * HID + 255) / 256, 256, 0, stream>>>(Wf1, Wf1b, HID * HID);
  cvt_bf16<<<(HID * HID + 255) / 256, 256, 0, stream>>>(Wf2, Wf2b, HID * HID);
  cvt_bf16<<<(int)(((size_t)HL * HID + 255) / 256), 256, 0, stream>>>(Wf3, Wf3b, HL * HID);
  cvt_bf16<<<(10 * HID + 255) / 256, 256, 0, stream>>>(Wr, WrBF, 10 * HID);
  zero_u32<<<1, 1, 0, stream>>>(cnt);

  // 2) parallel log-signatures
  logsig_kernel<<<(NSTEP + 255) / 256, 256, 0, stream>>>(values, logsigs);

  // 3) init MLP -> h0
  init_h0<<<1, HID, 0, stream>>>(values, Wi1, bi1, Wi2, bi2, Wi3, bi3,
                                 hiddenF, hiddenBF);

  // 4) sequential scan: one cluster of 16 WGPs, one sync per step
  scan_kernel<<<NWG, 512, 0, stream>>>(Wf1b, Wf2b, Wf3b, bf1, bf2, bf3,
                                       logsigs, hiddenF, hiddenBF, cnt);

  // 5) epilogue GEMM
  out_gemm<<<32, 256, 0, stream>>>(hiddenBF, WrBF, br, out);
}
